// GCNNet_33440615366817
// MI455X (gfx1250) — matrix operand order repaired and checked by reading, not compiled, not run on hardware
//
#include <hip/hip_runtime.h>
#include <hip/hip_bf16.h>

// ---------------------------------------------------------------------------
// GCN forward on gfx1250 (MI455X), split-precision bf16x3 WMMA GEMMs with all
// f32->bf16(hi,lo) conversion hoisted out of the GEMM inner loop, fused
// CSR-segment-sum + LayerNorm (+ReLU) that emits the next layer's bf16 inputs.
// ---------------------------------------------------------------------------

typedef __attribute__((ext_vector_type(16))) __bf16 v16bf;
typedef __attribute__((ext_vector_type(8)))  float  v8f;

__device__ __forceinline__ void split_bf16(float x, unsigned short& hi, unsigned short& lo) {
    unsigned u = __float_as_uint(x);
    hi = (unsigned short)(u >> 16);                       // truncated bf16 (exact split)
    float hif = __uint_as_float(u & 0xFFFF0000u);
    float lof = x - hif;                                  // exact residual
    lo = (unsigned short)(__float_as_uint(lof) >> 16);    // next 8 mantissa bits
}

// Load 16 bf16 = two contiguous 16-byte chunks into one v16bf fragment.
__device__ __forceinline__ v16bf load_frag(const unsigned short* __restrict__ p0,
                                           const unsigned short* __restrict__ p1) {
    struct P { uint4 a, b; } t;
    t.a = *(const uint4*)p0;
    t.b = *(const uint4*)p1;
    return __builtin_bit_cast(v16bf, t);
}

// One wave32 per 16x16 C tile. A (hi/lo): [M,K] row-major bf16.
// Bt (hi/lo): [DOUT,K] row-major bf16 (i.e. W transposed). C: [M,DOUT] f32.
template <int DOUT, int K>
__global__ __launch_bounds__(256) void gemm_wmma_bf16x3(
    const unsigned short* __restrict__ Ahi, const unsigned short* __restrict__ Alo,
    const unsigned short* __restrict__ Bthi, const unsigned short* __restrict__ Btlo,
    float* __restrict__ C, int Mrows) {
    const int wave = blockIdx.x * (blockDim.x >> 5) + (threadIdx.x >> 5);
    const int lane = threadIdx.x & 31;
    constexpr int TN = DOUT / 16;
    const int mt = wave / TN;
    const int nt = wave - mt * TN;
    const int m0 = mt * 16, n0 = nt * 16;
    if (m0 >= Mrows) return;                  // wave-uniform: EXEC all-ones below

    const int half = lane >> 4;               // 0: lanes 0-15, 1: lanes 16-31
    const int l    = lane & 15;

    int arow_i = m0 + l;
    if (arow_i >= Mrows) arow_i = Mrows - 1;  // safety (N is a multiple of 16 here)
    const unsigned short* __restrict__ ah = Ahi + (size_t)arow_i * K;
    const unsigned short* __restrict__ al = Alo + (size_t)arow_i * K;
    const unsigned short* __restrict__ bh = Bthi + (size_t)(n0 + l) * K;
    const unsigned short* __restrict__ bl = Btlo + (size_t)(n0 + l) * K;

    v8f acc = {};
#pragma unroll
    for (int kb = 0; kb < K; kb += 32) {
        // A fragment (ISA 7.12.2, 16-bit A 16x32): K = c0..c0+7 and c0+16..c0+23
        const int c0 = kb + half * 8;
        v16bf Ah = load_frag(ah + c0, ah + c0 + 16);
        v16bf Al = load_frag(al + c0, al + c0 + 16);
        // B fragment (16-bit B 32x16): column n0+l, K = r0..r0+7 and r0+16..r0+23
        const int r0 = kb + half * 8;
        v16bf Bh = load_frag(bh + r0, bh + r0 + 16);
        v16bf Bl = load_frag(bl + r0, bl + r0 + 16);
        // bf16x3: hi*hi + hi*lo + lo*hi  (~f32-class precision, f32 accumulate)
        acc = __builtin_amdgcn_wmma_f32_16x16x32_bf16(false, Ah, false, Bh, (short)0, acc, false, false);
        acc = __builtin_amdgcn_wmma_f32_16x16x32_bf16(false, Ah, false, Bl, (short)0, acc, false, false);
        acc = __builtin_amdgcn_wmma_f32_16x16x32_bf16(false, Al, false, Bh, (short)0, acc, false, false);
    }

    // C store (f32 16x16 layout): VGPR r -> row m0+half*8+r, lane -> col n0+l
    const int rbase = m0 + half * 8;
#pragma unroll
    for (int r = 0; r < 8; ++r) {
        int row = rbase + r;
        if (row < Mrows) C[(size_t)row * DOUT + n0 + l] = acc[r];
    }
}

// W [K,DOUT] f32  ->  Wt_hi / Wt_lo [DOUT,K] bf16 (transposed, split)
__global__ void convert_w(const float* __restrict__ W,
                          unsigned short* __restrict__ wt_hi,
                          unsigned short* __restrict__ wt_lo, int K, int DOUT) {
    int idx = blockIdx.x * blockDim.x + threadIdx.x;
    if (idx >= K * DOUT) return;
    int k = idx / DOUT, n = idx - k * DOUT;
    unsigned short h_, l_;
    split_bf16(W[idx], h_, l_);
    wt_hi[(size_t)n * K + k] = h_;
    wt_lo[(size_t)n * K + k] = l_;
}

// X [count] f32 -> hi/lo bf16 (same layout)
__global__ void convert_x(const float* __restrict__ X,
                          unsigned short* __restrict__ hi,
                          unsigned short* __restrict__ lo, int count) {
    int idx = blockIdx.x * blockDim.x + threadIdx.x;
    if (idx >= count) return;
    unsigned short h_, l_;
    split_bf16(X[idx], h_, l_);
    hi[idx] = h_;
    lo[idx] = l_;
}

// Fused CSR segment-sum + LayerNorm (+ReLU). One wave32 per node row.
// BF16OUT: emit bf16 hi/lo (next layer's GEMM input). Else: f32 to `out`.
template <int D, bool RELU, bool BF16OUT>
__global__ __launch_bounds__(256) void agg_ln(
    const float* __restrict__ xw, const int* __restrict__ rowptr,
    const int* __restrict__ cols, float* __restrict__ out,
    unsigned short* __restrict__ out_hi, unsigned short* __restrict__ out_lo,
    int nnodes) {
    const int node = blockIdx.x * (blockDim.x >> 5) + (threadIdx.x >> 5);
    if (node >= nnodes) return;
    const int lane = threadIdx.x & 31;
    constexpr int V = D / 32;                 // 4 (D=128) or 2 (D=64) floats/lane

    float acc[V];
#pragma unroll
    for (int i = 0; i < V; ++i) acc[i] = 0.0f;

    const int e0 = rowptr[node];
    const int e1 = rowptr[node + 1];
    for (int e = e0; e < e1; ++e) {
        const int src = cols[e];
        const float* __restrict__ p = xw + (size_t)src * D + lane * V;
        if constexpr (V == 4) {
            float4 v = *(const float4*)p;
            acc[0] += v.x; acc[1] += v.y; acc[2] += v.z; acc[3] += v.w;
        } else {
            float2 v = *(const float2*)p;
            acc[0] += v.x; acc[1] += v.y;
        }
    }

    float s = 0.0f, sq = 0.0f;
#pragma unroll
    for (int i = 0; i < V; ++i) { s += acc[i]; sq += acc[i] * acc[i]; }
#pragma unroll
    for (int off = 16; off > 0; off >>= 1) {
        s  += __shfl_xor(s,  off, 32);
        sq += __shfl_xor(sq, off, 32);
    }
    const float mu  = s * (1.0f / D);
    const float var = sq * (1.0f / D) - mu * mu;
    const float inv = rsqrtf(var + 1e-5f);

    if constexpr (BF16OUT) {
        unsigned int hp[V / 2], lp[V / 2];
#pragma unroll
        for (int j = 0; j < V / 2; ++j) {
            unsigned short h0, l0, h1, l1;
            float v0 = (acc[2 * j]     - mu) * inv;
            float v1 = (acc[2 * j + 1] - mu) * inv;
            if (RELU) { v0 = fmaxf(v0, 0.0f); v1 = fmaxf(v1, 0.0f); }
            split_bf16(v0, h0, l0);
            split_bf16(v1, h1, l1);
            hp[j] = (unsigned)h0 | ((unsigned)h1 << 16);
            lp[j] = (unsigned)l0 | ((unsigned)l1 << 16);
        }
        unsigned int* __restrict__ oh = (unsigned int*)(out_hi + (size_t)node * D + lane * V);
        unsigned int* __restrict__ ol = (unsigned int*)(out_lo + (size_t)node * D + lane * V);
#pragma unroll
        for (int j = 0; j < V / 2; ++j) { oh[j] = hp[j]; ol[j] = lp[j]; }
    } else {
        float* __restrict__ o = out + (size_t)node * D + lane * V;
#pragma unroll
        for (int i = 0; i < V; ++i) {
            float v = (acc[i] - mu) * inv;
            if (RELU) v = fmaxf(v, 0.0f);
            o[i] = v;
        }
    }
}

extern "C" void kernel_launch(void* const* d_in, const int* in_sizes, int n_in,
                              void* d_out, int out_size, void* d_ws, size_t ws_size,
                              hipStream_t stream) {
    const float* feat   = (const float*)d_in[0];
    const float* W0     = (const float*)d_in[1];
    const float* W1     = (const float*)d_in[2];
    const float* W2     = (const float*)d_in[3];
    const int*   rowptr = (const int*)d_in[4];    // nodePointer [N+1]
    const int*   cols   = (const int*)d_in[5];    // edgeList [E]
    (void)d_in[6]; (void)n_in; (void)out_size; (void)ws_size;

    const int N = in_sizes[4] - 1;                // 50000
    constexpr int DIN = 128, DH = 128, DOUT = 64;

    // Workspace layout
    char* ws = (char*)d_ws;
    float*          xw   = (float*)ws;                         ws += (size_t)N * DH * sizeof(float);   // 25.6 MB
    unsigned short* h_hi = (unsigned short*)ws;                ws += (size_t)N * DH * sizeof(short);   // 12.8 MB
    unsigned short* h_lo = (unsigned short*)ws;                ws += (size_t)N * DH * sizeof(short);   // 12.8 MB
    unsigned short* wt_hi = (unsigned short*)ws;               ws += (size_t)DH * DH * sizeof(short);  // 32 KB
    unsigned short* wt_lo = (unsigned short*)ws;

    float* out = (float*)d_out;

    const int mtiles = (N + 15) / 16;
    auto gemm_blocks = [&](int dout) { return (mtiles * (dout / 16) + 7) / 8; };
    const int agg_blocks = (N + 7) / 8;

    // ---- Layer 0: xw = feat @ W0 ; h = relu(LN(A @ xw)) (as bf16 hi/lo) ----
    convert_x<<<((N * DIN) + 255) / 256, 256, 0, stream>>>(feat, h_hi, h_lo, N * DIN);
    convert_w<<<((DIN * DH) + 255) / 256, 256, 0, stream>>>(W0, wt_hi, wt_lo, DIN, DH);
    gemm_wmma_bf16x3<DH, DIN><<<gemm_blocks(DH), 256, 0, stream>>>(h_hi, h_lo, wt_hi, wt_lo, xw, N);
    agg_ln<DH, true, true><<<agg_blocks, 256, 0, stream>>>(xw, rowptr, cols, nullptr, h_hi, h_lo, N);

    // ---- Layer 1 ----
    convert_w<<<((DH * DH) + 255) / 256, 256, 0, stream>>>(W1, wt_hi, wt_lo, DH, DH);
    gemm_wmma_bf16x3<DH, DH><<<gemm_blocks(DH), 256, 0, stream>>>(h_hi, h_lo, wt_hi, wt_lo, xw, N);
    agg_ln<DH, true, true><<<agg_blocks, 256, 0, stream>>>(xw, rowptr, cols, nullptr, h_hi, h_lo, N);

    // ---- Layer 2 (no ReLU, f32 output) ----
    convert_w<<<((DH * DOUT) + 255) / 256, 256, 0, stream>>>(W2, wt_hi, wt_lo, DH, DOUT);
    gemm_wmma_bf16x3<DOUT, DH><<<gemm_blocks(DOUT), 256, 0, stream>>>(h_hi, h_lo, wt_hi, wt_lo, xw, N);
    agg_ln<DOUT, false, false><<<agg_blocks, 256, 0, stream>>>(xw, rowptr, cols, out, nullptr, nullptr, N);
}
